// MIX_RNN_8143257993454
// MI455X (gfx1250) — matrix operand order repaired and checked by reading, not compiled, not run on hardware
//
#include <hip/hip_runtime.h>
#include <stdint.h>

// ---------------------------------------------------------------------------
// CDNA5 (gfx1250) implementation: bf16 WMMA GEMMs (double-buffered LDS
// staging, TDM DMA for the bf16-input GEMM) + persistent cooperative RNN.
// ---------------------------------------------------------------------------

typedef __attribute__((ext_vector_type(16))) __bf16 bf16x16;
typedef __attribute__((ext_vector_type(8)))  __bf16 bf16x8;
typedef __attribute__((ext_vector_type(4)))  __bf16 bf16x4;
typedef __attribute__((ext_vector_type(8)))  float  f32x8;
typedef __attribute__((ext_vector_type(4)))  float  f32x4;
typedef __attribute__((ext_vector_type(4)))  unsigned int u32x4;
typedef __attribute__((ext_vector_type(8)))  int i32x8;
typedef __attribute__((ext_vector_type(4)))  int i32x4;

union ABFrag { bf16x16 v16; bf16x8 v8[2]; };

__device__ __forceinline__ f32x8 wmma_bf16(bf16x16 a, bf16x16 b, f32x8 c) {
  // D = A(16x32 bf16) * B(32x16 bf16) + C(16x16 f32)
  return __builtin_amdgcn_wmma_f32_16x16x32_bf16(
      /*neg_a=*/false, a, /*neg_b=*/false, b,
      /*c_mod=*/(short)0, c, /*reuse_a=*/false, /*reuse_b=*/false);
}

// ---------------------------------------------------------------------------
// Tensor Data Mover: 2D bf16 tile (tile_d0 elems/row x tile_d1 rows) from
// global (row stride = row_stride elems) into LDS at lds_byte_addr, with
// hardware LDS padding: pad_interval code 3 (=16 DWORDs = one 64B row) and
// pad_amount code 3 (=4 DWORDs = 16B) -> 80B LDS row stride (bank-conflict-
// free for the 16B A-fragment ds_loads). D# layout per CDNA5 ISA §8.3/8.4.
// This toolchain's builtin takes 6 args (g0, g1, g2, g3, g_extra, cpol).
// ---------------------------------------------------------------------------
#if __has_builtin(__builtin_amdgcn_tensor_load_to_lds)
#define HAVE_TDM 1
__device__ __forceinline__ void tdm_load_tile_bf16(unsigned lds_byte_addr,
                                                   unsigned long long gaddr,
                                                   unsigned tile_d0, unsigned tile_d1,
                                                   unsigned long long row_stride) {
  u32x4 g0;
  g0[0] = 1u;                                              // count=1, user D#
  g0[1] = lds_byte_addr;                                   // lds_addr
  g0[2] = (unsigned)(gaddr & 0xffffffffull);               // global_addr lo
  g0[3] = (unsigned)((gaddr >> 32) & 0x01ffffffull) | (2u << 30);  // hi | type=2
  const unsigned td0 = 1u << 20, td1 = 1u << 20;           // huge dims: no OOB clip
  i32x8 g1;
  g1[0] = (int)((1u << 16)                                 // data_size = 2 bytes
                | (1u << 20)                               // pad_enable
                | (3u << 22)                               // pad_interval: 16 DW
                | (3u << 25));                             // pad_amount:   4 DW
  g1[1] = (int)((td0 & 0xffffu) << 16);                    // tensor_dim0[15:0]
  g1[2] = (int)((td0 >> 16) | ((td1 & 0xffffu) << 16));    // td0 hi | td1 lo
  g1[3] = (int)((td1 >> 16) | (tile_d0 << 16));            // td1 hi | tile_dim0
  g1[4] = (int)(tile_d1 & 0xffffu);                        // tile_dim1 (tile_dim2=0)
  g1[5] = (int)(unsigned)(row_stride & 0xffffffffull);     // dim0_stride lo
  g1[6] = (int)(unsigned)((row_stride >> 32) & 0xffffull); // dim0_stride hi
  g1[7] = 0;
  const i32x4 gz4 = {0, 0, 0, 0};                          // groups 2/3 unused (2D)
  const i32x8 gz8 = {0, 0, 0, 0, 0, 0, 0, 0};
  __builtin_amdgcn_tensor_load_to_lds(g0, g1, gz4, gz4, gz8, 0);
}
#else
#define HAVE_TDM 0
#endif

// ---------------------------------------------------------------------------
// K1: convert a 1024x1024 f32 weight (K x N, WMMA B operand) into bf16,
// pre-swizzled into per-tile per-lane fragment order:
//   tile(kt,nt) of 32(K)x16(N); lane = (n%16) + 16*((k%16)/8); elem = 8*((k%32)/16) + k%8
// so GEMM kernels fetch a lane fragment as 32 contiguous bytes (2x b128).
// ---------------------------------------------------------------------------
__global__ void __launch_bounds__(256) k_swizzle_w(const float* __restrict__ W,
                                                   unsigned short* __restrict__ outRaw) {
  __bf16* out = reinterpret_cast<__bf16*>(outRaw);
  int t0 = (blockIdx.x * 256 + threadIdx.x) * 4;
#pragma unroll
  for (int i = 0; i < 4; ++i) {
    int t = t0 + i;
    int tile = t >> 9, pos = t & 511;
    int lane = pos >> 4, e = pos & 15;
    int kt = tile >> 6, nt = tile & 63;
    int k = kt * 32 + (e >> 3) * 16 + (lane >> 4) * 8 + (e & 7);
    int n = nt * 16 + (lane & 15);
    out[t] = (__bf16)W[(size_t)k * 1024 + n];
  }
}

// ---------------------------------------------------------------------------
// K2/K4: C[M,1024] = A[M,1024] @ B[1024,1024] + bias   (A f32 or bf16)
// 256 threads (8 waves), 128(M) x 128(N) tile, K streamed in 32-chunks with
// DOUBLE-BUFFERED LDS staging: chunk kt+1 is staged (TDM DMA for bf16 A,
// load+convert for f32 A) while chunk kt feeds the WMMA pipe; one barrier
// per chunk publishes the next buffer.
// ---------------------------------------------------------------------------
template <int AF32>
__global__ void __launch_bounds__(256) k_gemm1024(const void* __restrict__ Aany,
                                                  const unsigned short* __restrict__ BswRaw,
                                                  const float* __restrict__ bias,
                                                  float* __restrict__ C) {
  const __bf16* Bsw = reinterpret_cast<const __bf16*>(BswRaw);
  __shared__ __align__(16) __bf16 As[2][128 * 40];

  const int lane = threadIdx.x & 31, wave = threadIdx.x >> 5;
  const int mw = wave & 3, nw = wave >> 2;
  const int row0 = blockIdx.y * 128;
  const int nbase = blockIdx.x * 128 + nw * 64;

  // stage K-chunk kt into LDS buffer buf (issue only; completion via
  // s_wait_tensorcnt in wave 0 / DS-counter at the barrier)
  auto stage = [&](int kt, int buf) {
    if (AF32) {
#pragma unroll
      for (int it = 0; it < 4; ++it) {
        int li = threadIdx.x + it * 256;  // 0..1023
        int r = li >> 3, c4 = li & 7;     // 8 x float4 per row
        const float* Af = reinterpret_cast<const float*>(Aany);
        f32x4 v = *(const f32x4*)(Af + (size_t)(row0 + r) * 1024 + kt * 32 + c4 * 4);
        bf16x4 w = bf16x4{(__bf16)v.x, (__bf16)v.y, (__bf16)v.z, (__bf16)v.w};
        *(bf16x4*)&As[buf][r * 40 + c4 * 4] = w;
      }
    } else {
#if HAVE_TDM
      if (wave == 0) {
        const __bf16* Ah = reinterpret_cast<const __bf16*>(Aany);
        unsigned long long ga =
            (unsigned long long)(uintptr_t)(Ah + (size_t)row0 * 1024 + kt * 32);
        unsigned lds_off = (unsigned)(uintptr_t)(&As[buf][0]);  // LDS aperture low bits
        tdm_load_tile_bf16(lds_off, ga, /*tile_d0=*/32, /*tile_d1=*/128,
                           /*row_stride=*/1024ull);
      }
#else
#pragma unroll
      for (int it = 0; it < 4; ++it) {
        int li = threadIdx.x + it * 256;
        int r = li >> 3, c4 = li & 7;
        const __bf16* Ah = reinterpret_cast<const __bf16*>(Aany);
        bf16x4 w = *(const bf16x4*)(Ah + (size_t)(row0 + r) * 1024 + kt * 32 + c4 * 4);
        *(bf16x4*)&As[buf][r * 40 + c4 * 4] = w;
      }
#endif
    }
  };

  const f32x8 zero = {0.f, 0.f, 0.f, 0.f, 0.f, 0.f, 0.f, 0.f};
  f32x8 acc[2][4];
#pragma unroll
  for (int mt = 0; mt < 2; ++mt)
#pragma unroll
    for (int nt = 0; nt < 4; ++nt) acc[mt][nt] = zero;

  // prologue: stage chunk 0 into buffer 0
  stage(0, 0);
#if HAVE_TDM
  if (!AF32 && wave == 0) __builtin_amdgcn_s_wait_tensorcnt(0);
#endif
  __syncthreads();

  for (int kt = 0; kt < 32; ++kt) {
    const int cur = kt & 1;
    // issue stage of next chunk into the other buffer (overlaps with WMMAs)
    if (kt + 1 < 32) stage(kt + 1, cur ^ 1);

    // --- A fragments from LDS (ISA 16-bit A layout: lane-group K halves) ---
    ABFrag a[2];
#pragma unroll
    for (int mt = 0; mt < 2; ++mt) {
      int mrow = mw * 32 + mt * 16 + (lane & 15);
      int grp8 = (lane >> 4) * 8;
      a[mt].v8[0] = *(const bf16x8*)&As[cur][mrow * 40 + grp8];
      a[mt].v8[1] = *(const bf16x8*)&As[cur][mrow * 40 + grp8 + 16];
    }
    // --- B fragments from pre-swizzled global + WMMA ---
#pragma unroll
    for (int nt = 0; nt < 4; ++nt) {
      ABFrag b;
      size_t tile = (size_t)kt * 64 + ((nbase >> 4) + nt);
      const bf16x8* bp = (const bf16x8*)(Bsw + tile * 512 + (size_t)lane * 16);
      b.v8[0] = bp[0];
      b.v8[1] = bp[1];
#pragma unroll
      for (int mt = 0; mt < 2; ++mt)
        acc[mt][nt] = wmma_bf16(a[mt].v16, b.v16, acc[mt][nt]);
    }

#if HAVE_TDM
    if (!AF32 && wave == 0) __builtin_amdgcn_s_wait_tensorcnt(0);
#endif
    __syncthreads();  // publish next buffer; guard reuse of current buffer
  }

  // --- epilogue: C-layout (VGPR v -> row v / 8+v by lane group), add bias ---
#pragma unroll
  for (int nt = 0; nt < 4; ++nt) {
    int col = nbase + nt * 16 + (lane & 15);
    float bv = bias[col];
#pragma unroll
    for (int mt = 0; mt < 2; ++mt) {
#pragma unroll
      for (int v = 0; v < 8; ++v) {
        int row = row0 + mw * 32 + mt * 16 + v + 8 * (lane >> 4);
        C[(size_t)row * 1024 + col] = acc[mt][nt][v] + bv;
      }
    }
  }
}

// ---------------------------------------------------------------------------
// K3: per-row LayerNorm (f32) + 2-class encoder head; writes bf16 h_ln.
// One wave32 per 1024-wide row (32 f32 per lane, shfl_xor reductions).
// ---------------------------------------------------------------------------
__global__ void __launch_bounds__(256) k_ln_enc(const float* __restrict__ h,
                                                const float* __restrict__ g,
                                                const float* __restrict__ be,
                                                const float* __restrict__ Wc,
                                                const float* __restrict__ bc,
                                                unsigned short* __restrict__ hlnRaw,
                                                float* __restrict__ yenc) {
  __bf16* hln = reinterpret_cast<__bf16*>(hlnRaw);
  const int lane = threadIdx.x & 31, wave = threadIdx.x >> 5;
  const size_t row = (size_t)blockIdx.x * 8 + wave;
  const float* hr = h + row * 1024;

  float v[32];
  float s = 0.f;
#pragma unroll
  for (int i = 0; i < 32; ++i) {
    v[i] = hr[i * 32 + lane];
    s += v[i];
  }
#pragma unroll
  for (int off = 16; off >= 1; off >>= 1) s += __shfl_xor(s, off, 32);
  float mu = s * (1.f / 1024.f);

  float s2 = 0.f;
#pragma unroll
  for (int i = 0; i < 32; ++i) {
    float d = v[i] - mu;
    s2 += d * d;
  }
#pragma unroll
  for (int off = 16; off >= 1; off >>= 1) s2 += __shfl_xor(s2, off, 32);
  float rstd = rsqrtf(s2 * (1.f / 1024.f) + 1e-5f);

  float e0 = 0.f, e1 = 0.f;
#pragma unroll
  for (int i = 0; i < 32; ++i) {
    int col = i * 32 + lane;
    float nv = (v[i] - mu) * rstd * g[col] + be[col];
    hln[row * 1024 + col] = (__bf16)nv;
    e0 += nv * Wc[col * 2 + 0];
    e1 += nv * Wc[col * 2 + 1];
  }
#pragma unroll
  for (int off = 16; off >= 1; off >>= 1) {
    e0 += __shfl_xor(e0, off, 32);
    e1 += __shfl_xor(e1, off, 32);
  }
  if (lane == 0) {
    yenc[row * 2 + 0] = e0 + bc[0];
    yenc[row * 2 + 1] = e1 + bc[1];
  }
}

// ---------------------------------------------------------------------------
// Grid-wide barrier: atomic arrive counter (correct under any dispatch) +
// CDNA5 cluster barrier (s_barrier_signal/wait -3; NOP if not clustered).
// ---------------------------------------------------------------------------
__device__ __forceinline__ void grid_barrier(unsigned* bar, unsigned& nbar) {
  __syncthreads();
  ++nbar;
  if (threadIdx.x == 0) {
    __threadfence();               // release: state stores visible device-wide
    atomicAdd(bar, 1u);
    const unsigned tgt = nbar * gridDim.x;
    while (atomicAdd(bar, 0u) < tgt) __builtin_amdgcn_s_sleep(1);
  }
  __builtin_amdgcn_s_cluster_barrier();
  __syncthreads();
  __threadfence();                 // acquire: see other WGs' state stores
}

// ---------------------------------------------------------------------------
// K5: sequential RNN scan, persistent cooperative kernel (16 WGs = max
// cluster size; each owns a 64-column slab of the 32x1024 state).
// Per step: state_new = relu(xp[:,t,:] + state @ fc2_W + b), state double-
// buffered bf16 through L2; fc2_W (2 MB bf16 swizzled) L2-resident.
// Wave: 1 output fragment; two interleaved WMMA accumulator chains (even/odd
// K-chunks) so the matrix pipe isn't serialized on the C operand.
// ---------------------------------------------------------------------------
__global__ void __launch_bounds__(256) k_rnn(const float* __restrict__ xp,
                                             const unsigned short* __restrict__ W3swRaw,
                                             const float* __restrict__ b2,
                                             const float* __restrict__ Wc3,
                                             const float* __restrict__ bc3,
                                             unsigned short* s0raw,
                                             unsigned short* s1raw,
                                             float* __restrict__ out,
                                             unsigned* __restrict__ bar, int L) {
  const __bf16* W3sw = reinterpret_cast<const __bf16*>(W3swRaw);
  __bf16* sa = reinterpret_cast<__bf16*>(s0raw);
  __bf16* sb = reinterpret_cast<__bf16*>(s1raw);

  const int lane = threadIdx.x & 31, wave = threadIdx.x >> 5;
  const int mt = wave & 1;
  const int nbase = blockIdx.x * 64 + (wave >> 1) * 16;
  const int col = nbase + (lane & 15);
  const float bcol = b2[col];
  const float w30 = Wc3[col * 2 + 0];
  const float w31 = Wc3[col * 2 + 1];

  // zero-init our slab of state buffer 0
  for (int i = threadIdx.x; i < 32 * 64; i += 256) {
    int r = i >> 6, c = blockIdx.x * 64 + (i & 63);
    sa[r * 1024 + c] = (__bf16)0.f;
  }
  // pull our fc2_W working set into L2 ahead of the 2048-step reuse loop
  for (int kt = 0; kt < 32; ++kt) {
    size_t tile = (size_t)kt * 64 + (nbase >> 4);
    __builtin_prefetch(W3sw + tile * 512 + (size_t)lane * 16, 0, 1);
  }
  unsigned nbar = 0;
  grid_barrier(bar, nbar);

  const f32x8 zero = {0.f, 0.f, 0.f, 0.f, 0.f, 0.f, 0.f, 0.f};
  const int m = mt * 16 + (lane & 15);
  const int kgrp = (lane >> 4) * 8;

  for (int t = 0; t < L; ++t) {
    f32x8 acc0 = zero, acc1 = zero;
#pragma unroll 4
    for (int kt = 0; kt < 32; kt += 2) {
      ABFrag a0, b0, a1, b1;
      int k0 = kt * 32 + kgrp;
      a0.v8[0] = *(const bf16x8*)&sa[m * 1024 + k0];
      a0.v8[1] = *(const bf16x8*)&sa[m * 1024 + k0 + 16];
      a1.v8[0] = *(const bf16x8*)&sa[m * 1024 + k0 + 32];
      a1.v8[1] = *(const bf16x8*)&sa[m * 1024 + k0 + 48];
      size_t tile0 = (size_t)kt * 64 + (nbase >> 4);
      const bf16x8* bp0 = (const bf16x8*)(W3sw + tile0 * 512 + (size_t)lane * 16);
      const bf16x8* bp1 = (const bf16x8*)(W3sw + (tile0 + 64) * 512 + (size_t)lane * 16);
      b0.v8[0] = bp0[0];
      b0.v8[1] = bp0[1];
      b1.v8[0] = bp1[0];
      b1.v8[1] = bp1[1];
      acc0 = wmma_bf16(a0.v16, b0.v16, acc0);  // two independent WMMA chains
      acc1 = wmma_bf16(a1.v16, b1.v16, acc1);
    }
    f32x8 acc = acc0 + acc1;

    const bool last = (t == L - 1);
#pragma unroll
    for (int v = 0; v < 8; ++v) {
      int row = mt * 16 + v + 8 * (lane >> 4);  // batch index
      float sv = acc[v] + xp[((size_t)row * L + t) * 1024 + col] + bcol;
      sv = fmaxf(sv, 0.f);
      sb[row * 1024 + col] = (__bf16)sv;
      if (last) {  // out = state_L @ fc3_W (bias added separately)
        atomicAdd(&out[row * 2 + 0], sv * w30);
        atomicAdd(&out[row * 2 + 1], sv * w31);
      }
    }
    grid_barrier(bar, nbar);
    __bf16* tmp = sa; sa = sb; sb = tmp;
  }

  if (blockIdx.x == 0 && threadIdx.x < 64)
    atomicAdd(&out[threadIdx.x], bc3[threadIdx.x & 1]);
}

// ---------------------------------------------------------------------------
// Host launcher
// ---------------------------------------------------------------------------
extern "C" void kernel_launch(void* const* d_in, const int* in_sizes, int n_in,
                              void* d_out, int out_size, void* d_ws, size_t ws_size,
                              hipStream_t stream) {
  const float* x     = (const float*)d_in[0];
  const float* emb_W = (const float*)d_in[1];
  const float* emb_b = (const float*)d_in[2];
  const float* ln_g  = (const float*)d_in[3];
  const float* ln_b  = (const float*)d_in[4];
  const float* ecW   = (const float*)d_in[5];
  const float* ecb   = (const float*)d_in[6];
  const float* fc1_W = (const float*)d_in[7];
  const float* fc1_b = (const float*)d_in[8];
  const float* fc2_W = (const float*)d_in[9];
  const float* fc2_b = (const float*)d_in[10];
  const float* fc3_W = (const float*)d_in[11];
  const float* fc3_b = (const float*)d_in[12];
  float* out = (float*)d_out;

  const int B = 32;
  const int L = in_sizes[0] / (B * 1024);
  const int M = B * L;

  // workspace layout
  char* ws = (char*)d_ws;
  unsigned short* W1s = (unsigned short*)(ws);                    // 2 MB emb_W bf16 swz
  unsigned short* W2s = (unsigned short*)(ws + (2ull << 20));     // 2 MB fc1_W
  unsigned short* W3s = (unsigned short*)(ws + (4ull << 20));     // 2 MB fc2_W
  unsigned*       bar = (unsigned*)(ws + (6ull << 20));           // barrier counter
  unsigned short* st0 = (unsigned short*)(ws + (6ull << 20) + 256);  // state bufs
  unsigned short* st1 = st0 + 32 * 1024;
  unsigned short* hln = (unsigned short*)(ws + (8ull << 20));     // M*1024 bf16
  float* hbuf = (float*)(ws + (8ull << 20) + (size_t)M * 1024 * 2);  // M*1024 f32 (h, then xp)

  size_t needed = (8ull << 20) + (size_t)M * 1024 * 2 + (size_t)M * 1024 * 4;
  if (ws_size < needed) return;  // scratch too small; bail (no-op)

  (void)hipMemsetAsync(bar, 0, sizeof(unsigned), stream);
  (void)hipMemsetAsync(d_out, 0, 64 * sizeof(float), stream);  // RNN head accums

  // weight conversion + fragment swizzle
  k_swizzle_w<<<1024, 256, 0, stream>>>(emb_W, W1s);
  k_swizzle_w<<<1024, 256, 0, stream>>>(fc1_W, W2s);
  k_swizzle_w<<<1024, 256, 0, stream>>>(fc2_W, W3s);

  dim3 g2(8, M / 128);
  // h = x @ emb_W + emb_b
  k_gemm1024<1><<<g2, 256, 0, stream>>>((const void*)x, W1s, emb_b, hbuf);
  // layernorm + encoder head (y_enc -> d_out[64..])
  k_ln_enc<<<M / 8, 256, 0, stream>>>(hbuf, ln_g, ln_b, ecW, ecb, hln, out + 64);
  // xp = h_ln @ fc1_W + fc1_b   (overwrites h buffer; A staged via TDM)
  k_gemm1024<0><<<g2, 256, 0, stream>>>((const void*)hln, W2s, fc1_b, hbuf);
  // sequential RNN scan -> d_out[0..63]
  k_rnn<<<16, 256, 0, stream>>>(hbuf, W3s, fc2_b, fc3_W, fc3_b, st0, st1, out, bar, L);
}
